// SubMConv3dTestTorch_83794811945697
// MI455X (gfx1250) — matrix operand order repaired and checked by reading, not compiled, not run on hardware
//
#include <hip/hip_runtime.h>

typedef __attribute__((ext_vector_type(16))) __bf16 v16bf;
typedef __attribute__((ext_vector_type(8)))  float  v8f;

// exact pointee type expected by the async-LDS builtins (from the clang diagnostic)
typedef int b128_t __attribute__((vector_size(16)));
typedef __attribute__((address_space(1))) b128_t* gas_b128;   // global (__device__) ptr
typedef __attribute__((address_space(3))) b128_t* las_b128;   // LDS (__shared__) ptr

#define C 64
#define KVOL 27
#define DIM_D 41
#define DIM_H 1024
#define DIM_W 1024
#define TS (1 << 18)          // hash slots (>= 2x N), power of two
#define TSM (TS - 1)

#define WROW_B 144            // padded LDS row stride: 64 bf16 = 128B + 16B pad
#define WBUF_B (64 * WROW_B)  // one tap's weight tile in LDS (9216 B)

__device__ __forceinline__ unsigned hash_lin(int lin) {
    return ((unsigned)lin * 2654435761u) & TSM;
}

__device__ __forceinline__ __bf16 f2bf(float f) {
    union { float f; unsigned u; } in; in.f = f;
    unsigned u = in.u;
    u = (u + 0x7FFFu + ((u >> 16) & 1u)) >> 16;   // round-to-nearest-even
    union { unsigned short s; __bf16 b; } out; out.s = (unsigned short)u;
    return out.b;
}

// ---- CDNA5 async global->LDS copy (ASYNCcnt-tracked) ----
__device__ __forceinline__ void async_copy_b128(const void* gsrc, void* ldst) {
#if __has_builtin(__builtin_amdgcn_global_load_async_to_lds_b128)
    __builtin_amdgcn_global_load_async_to_lds_b128(
        (gas_b128)gsrc, (las_b128)ldst, 0, 0);
#else
    unsigned loff = (unsigned)(unsigned long long)ldst;  // LDS offset = low 32 bits
    asm volatile("global_load_async_to_lds_b128 %0, %1, off"
                 :: "v"(loff), "v"(gsrc) : "memory");
#endif
}

#if __has_builtin(__builtin_amdgcn_s_wait_asynccnt)
#define WAIT_ASYNC(n_) __builtin_amdgcn_s_wait_asynccnt(n_)
#else
#define WAIT_ASYNC(n_) asm volatile("s_wait_asynccnt %0" :: "n"(n_) : "memory")
#endif

// ---------------- phase 1: hash grid ----------------
__global__ __launch_bounds__(256) void init_hash(int* keys) {
    int i = blockIdx.x * blockDim.x + threadIdx.x;
    if (i < TS) keys[i] = -1;
}

__global__ __launch_bounds__(256) void insert_hash(const int* __restrict__ coors, int n,
                                                   int* __restrict__ keys, int* __restrict__ vals) {
    int i = blockIdx.x * blockDim.x + threadIdx.x;
    if (i >= n) return;
    int b = coors[i * 4 + 0], z = coors[i * 4 + 1], y = coors[i * 4 + 2], x = coors[i * 4 + 3];
    int lin = ((b * DIM_D + z) * DIM_H + y) * DIM_W + x;
    unsigned slot = hash_lin(lin);
    for (;;) {
        int prev = atomicCAS(&keys[slot], -1, lin);
        if (prev == -1 || prev == lin) { vals[slot] = i; break; }
        slot = (slot + 1) & TSM;
    }
}

// ---------------- phase 2: rulebook rb[k*N + i] ----------------
__global__ __launch_bounds__(256) void build_rulebook(const int* __restrict__ coors, int n,
                                                      const int* __restrict__ keys,
                                                      const int* __restrict__ vals,
                                                      int* __restrict__ rb) {
    int t = blockIdx.x * blockDim.x + threadIdx.x;
    if (t >= n * KVOL) return;
    int k = t / n;
    int i = t - k * n;
    int dz = k / 9 - 1, dy = (k / 3) % 3 - 1, dx = k % 3 - 1;
    int b = coors[i * 4 + 0];
    int nz = coors[i * 4 + 1] + dz;
    int ny = coors[i * 4 + 2] + dy;
    int nx = coors[i * 4 + 3] + dx;
    int res = -1;
    if (nz >= 0 && nz < DIM_D && ny >= 0 && ny < DIM_H && nx >= 0 && nx < DIM_W) {
        int lin = ((b * DIM_D + nz) * DIM_H + ny) * DIM_W + nx;
        unsigned slot = hash_lin(lin);
        for (;;) {
            int key = keys[slot];
            if (key == lin) { res = vals[slot]; break; }
            if (key == -1) break;
            slot = (slot + 1) & TSM;
        }
    }
    rb[t] = res;
}

// ---------------- phase 3: weight transpose+convert  Wt[l][k][cout][cin] ----------------
__global__ __launch_bounds__(256) void prep_weights(const float* __restrict__ Ws, __bf16* __restrict__ Wt) {
    int t = blockIdx.x * blockDim.x + threadIdx.x;
    if (t >= 3 * KVOL * C * C) return;
    int cin = t & 63;
    int cout = (t >> 6) & 63;
    int lk = t >> 12;
    Wt[t] = f2bf(Ws[lk * C * C + cin * C + cout]);
}

__global__ __launch_bounds__(256) void feat_to_bf16(const float* __restrict__ f, __bf16* __restrict__ o,
                                                    __bf16* __restrict__ zrow, int total) {
    int t = blockIdx.x * blockDim.x + threadIdx.x;
    if (t < total) o[t] = f2bf(f[t]);
    if (t < C) zrow[t] = f2bf(0.0f);   // zero row for invalid-neighbor gathers
}

// ---------------- phase 4: gather-GEMM layer (WMMA bf16 -> f32) ----------------
// 256 threads = 8 waves; wave w owns rows [wave_base, wave_base+16) x all 64 cols.
// 3-deep software pipeline per tap t:
//   phase t-2: rulebook index loaded
//   phase t-1: A rows gathered (branchless; invalid -> zero row), weights async->LDS
//   phase t  : 8 x v_wmma_f32_16x16x32_bf16 from LDS B + register A
__global__ __launch_bounds__(256) void subm_layer_wmma(
    const __bf16* __restrict__ feat,   // N x 64 bf16
    const __bf16* __restrict__ zrow,   // 64 bf16 of zeros
    const __bf16* __restrict__ Wt,     // 27 x 64(cout) x 64(cin) bf16
    const int*    __restrict__ rb,     // 27 x N
    __bf16*       __restrict__ out_bf, // N x 64 (next-layer input)
    float*        __restrict__ out_f32,// N x 64 (final output)
    int n, int last) {

    __shared__ __align__(16) unsigned char wlds[2][WBUF_B];

    const int tid  = threadIdx.x;
    const int lane = tid & 31;
    const int wave = tid >> 5;
    const int half = lane >> 4;       // K-chunk selector for A/B fragments
    const int m    = lane & 15;       // A row within tile / B,C column
    const int wave_base = (blockIdx.x * 8 + wave) * 16;
    const int row  = wave_base + m;
    const int rowc = (row < n) ? row : (n - 1);   // clamped: rb reads unconditional

    union Frag { v16bf v; uint4 u[2]; };

    // Stage one tap's 64x64 bf16 weight tile (8 KB = 512 x 16B chunks, 2 per thread)
    // into padded LDS rows: chunk c -> row (c>>3), 16B-column (c&7).
    auto stage = [&](int k, int buf) {
        const char* src = (const char*)(Wt + (size_t)k * C * C);
        int c0 = tid, c1 = tid + 256;
        async_copy_b128(src + (size_t)c0 * 16, &wlds[buf][(c0 >> 3) * WROW_B + (c0 & 7) * 16]);
        async_copy_b128(src + (size_t)c1 * 16, &wlds[buf][(c1 >> 3) * WROW_B + (c1 & 7) * 16]);
    };

    // rulebook value for tap k (clamped so look-ahead never over-reads rb)
    auto rbv = [&](int k) {
        int kk = (k < KVOL) ? k : (KVOL - 1);
        return rb[kk * n + rowc];
    };

    // A fragments (16x32 bf16 each, kk = 0,1), contiguous layout matching B:
    //   element e -> cin = kk*32 + half*16 + e
    // Branchless: invalid rows (r < 0) gather from the zero row.
    auto loadA = [&](Frag& x0, Frag& x1, int r) {
        const uint4* fp = (r >= 0) ? reinterpret_cast<const uint4*>(feat + (size_t)r * C)
                                   : reinterpret_cast<const uint4*>(zrow);
        x0.u[0] = fp[2 * half];     x0.u[1] = fp[2 * half + 1];
        x1.u[0] = fp[4 + 2 * half]; x1.u[1] = fp[5 + 2 * half];
    };

    v8f acc0 = {}, acc1 = {}, acc2 = {}, acc3 = {};

    // One tap: 4 N-tiles x 2 K-steps = 8 WMMAs, B fragments from LDS
    // (lane = column cout, element e -> cin = kk*32 + half*16 + e).
    auto computeTap = [&](const unsigned char* wb, const Frag& x0, const Frag& x1) {
#pragma unroll
        for (int nt = 0; nt < 4; ++nt) {
            const unsigned char* wrow = wb + (nt * 16 + m) * WROW_B + half * 32;
            Frag b0, b1;
            b0.u[0] = *(const uint4*)(wrow);
            b0.u[1] = *(const uint4*)(wrow + 16);
            b1.u[0] = *(const uint4*)(wrow + 64);
            b1.u[1] = *(const uint4*)(wrow + 80);
            v8f* acc = (nt == 0) ? &acc0 : (nt == 1) ? &acc1 : (nt == 2) ? &acc2 : &acc3;
            *acc = __builtin_amdgcn_wmma_f32_16x16x32_bf16(false, x0.v, false, b0.v,
                                                           (short)0, *acc, false, false);
            *acc = __builtin_amdgcn_wmma_f32_16x16x32_bf16(false, x1.v, false, b1.v,
                                                           (short)0, *acc, false, false);
        }
    };

    // ---- prologue: tap0 A + weights; rb for tap1 in flight (slot S1)
    Frag aA0, aA1, aB0, aB1;
    stage(0, 0);
    int r0tmp = rbv(0);
    int S1 = rbv(1);          // rb value for tap 1 (used next phase)
    int S0;                   // rb slot written at even phases
    loadA(aA0, aA1, r0tmp);

    for (int k = 0; k < KVOL; k += 2) {
        // ---- even phase (tap k): compute (wlds[0], aA); prep tap k+1; rb for k+2 -> S0
        __syncthreads();                          // prior compute done; wlds[1] reusable
        if (k + 1 < KVOL) {
            stage(k + 1, 1);
            S0 = rbv(k + 2);                      // rb two taps ahead (hidden by this phase)
            loadA(aB0, aB1, S1);                  // rb value loaded a full phase ago
            WAIT_ASYNC(2);                        // tap-k staging complete (2 newer in flight)
        } else {
            WAIT_ASYNC(0);
        }
        __syncthreads();                          // wlds[0] visible to all waves
        computeTap(wlds[0], aA0, aA1);
        if (k + 1 >= KVOL) break;                 // KVOL odd: last tap is even

        // ---- odd phase (tap k+1): compute (wlds[1], aB); prep tap k+2; rb for k+3 -> S1
        __syncthreads();
        if (k + 2 < KVOL) {
            stage(k + 2, 0);
            S1 = rbv(k + 3);
            loadA(aA0, aA1, S0);
            WAIT_ASYNC(2);
        } else {
            WAIT_ASYNC(0);
        }
        __syncthreads();
        computeTap(wlds[1], aB0, aB1);
    }

    // C/D layout: VGPR j, lanes 0-15 -> M = j, lanes 16-31 -> M = 8+j; N = lane&15
#pragma unroll
    for (int nt = 0; nt < 4; ++nt) {
        const v8f acc = (nt == 0) ? acc0 : (nt == 1) ? acc1 : (nt == 2) ? acc2 : acc3;
        int col = nt * 16 + m;
#pragma unroll
        for (int j = 0; j < 8; ++j) {
            int orow = wave_base + half * 8 + j;
            if (orow < n) {
                if (last) out_f32[(size_t)orow * C + col] = acc[j];
                else      out_bf [(size_t)orow * C + col] = f2bf(acc[j]);
            }
        }
    }
}

// ---------------- host ----------------
static inline size_t align_up(size_t v, size_t a) { return (v + a - 1) & ~(a - 1); }

extern "C" void kernel_launch(void* const* d_in, const int* in_sizes, int n_in,
                              void* d_out, int out_size, void* d_ws, size_t ws_size,
                              hipStream_t stream) {
    const float* features = (const float*)d_in[0];
    const float* Ws       = (const float*)d_in[1];
    const int*   coors    = (const int*)d_in[2];
    float*       out      = (float*)d_out;
    const int n = in_sizes[0] / C;

    char* ws = (char*)d_ws;
    size_t off = 0;
    int* keys = (int*)(ws + off);       off = align_up(off + (size_t)TS * 4, 256);
    int* vals = (int*)(ws + off);       off = align_up(off + (size_t)TS * 4, 256);
    int* rb   = (int*)(ws + off);       off = align_up(off + (size_t)KVOL * n * 4, 256);
    __bf16* Wt = (__bf16*)(ws + off);   off = align_up(off + (size_t)3 * KVOL * C * C * 2, 256);
    __bf16* zrow = (__bf16*)(ws + off); off = align_up(off + (size_t)C * 2, 256);
    __bf16* f0 = (__bf16*)(ws + off);   off = align_up(off + (size_t)n * C * 2, 256);
    __bf16* f1 = (__bf16*)(ws + off);   off = align_up(off + (size_t)n * C * 2, 256);
    __bf16* f2 = (__bf16*)(ws + off);   off = align_up(off + (size_t)n * C * 2, 256);

    init_hash<<<TS / 256, 256, 0, stream>>>(keys);
    insert_hash<<<(n + 255) / 256, 256, 0, stream>>>(coors, n, keys, vals);
    build_rulebook<<<(KVOL * n + 255) / 256, 256, 0, stream>>>(coors, n, keys, vals, rb);
    prep_weights<<<(3 * KVOL * C * C + 255) / 256, 256, 0, stream>>>(Ws, Wt);
    feat_to_bf16<<<(n * C + 255) / 256, 256, 0, stream>>>(features, f0, zrow, n * C);

    const int gemm_blocks = (n + 127) / 128;
    subm_layer_wmma<<<gemm_blocks, 256, 0, stream>>>(f0, zrow, Wt + 0 * KVOL * C * C, rb, f1, out, n, 0);
    subm_layer_wmma<<<gemm_blocks, 256, 0, stream>>>(f1, zrow, Wt + 1 * KVOL * C * C, rb, f2, out, n, 0);
    subm_layer_wmma<<<gemm_blocks, 256, 0, stream>>>(f2, zrow, Wt + 2 * KVOL * C * C, rb, f0, out, n, 1);
}
